// WukongLayer_36318243455424
// MI455X (gfx1250) — matrix operand order repaired and checked by reading, not compiled, not run on hardware
//
#include <hip/hip_runtime.h>
#include <hip/hip_bf16.h>
#include <math.h>

#define B_SZ   2048
#define N_INP  64
#define D_     128
#define N_LCB  16
#define N_FMB  48
#define RANK   24
#define RANKP  32
#define MLP_IN 1536
#define DIMH   512
#define MLP_OUT 6144

typedef __bf16 bf16;
typedef __attribute__((ext_vector_type(16))) __bf16 v16bf;
typedef __attribute__((ext_vector_type(8)))  __bf16 v8bf;
typedef __attribute__((ext_vector_type(4)))  __bf16 v4bf;
typedef __attribute__((ext_vector_type(2)))  __bf16 v2bf;
typedef __attribute__((ext_vector_type(8)))  float  v8f;
typedef __attribute__((ext_vector_type(4)))  unsigned int uint4v;
typedef __attribute__((ext_vector_type(4)))  unsigned int u32x4;
typedef __attribute__((ext_vector_type(8)))  int i32x8;
typedef __attribute__((ext_vector_type(4)))  int i32x4;

union Frag {
    v16bf v;
    uint4v q[2];
};

__device__ __forceinline__ v8f wmma_bf16(v16bf a, v16bf b, v8f c) {
    return __builtin_amdgcn_wmma_f32_16x16x32_bf16(false, a, false, b,
                                                   (short)0, c, false, false);
}
__device__ __forceinline__ v8bf cvt8(float4 a, float4 b) {
    v8bf r;
    r[0] = (bf16)a.x; r[1] = (bf16)a.y; r[2] = (bf16)a.z; r[3] = (bf16)a.w;
    r[4] = (bf16)b.x; r[5] = (bf16)b.y; r[6] = (bf16)b.z; r[7] = (bf16)b.w;
    return r;
}
__device__ __forceinline__ v4bf cvt4(float4 a) {
    v4bf r;
    r[0] = (bf16)a.x; r[1] = (bf16)a.y; r[2] = (bf16)a.z; r[3] = (bf16)a.w;
    return r;
}

// ---------------------------------------------------------------------------
// Weight prep: W (K x N, f32) -> WT (N x K, bf16). 32x32 LDS tiles.
// ---------------------------------------------------------------------------
__global__ __launch_bounds__(256) void transpose_bf16_kernel(
    const float* __restrict__ W, bf16* __restrict__ WT, int K, int N)
{
    __shared__ float tile[32][33];
    const int k0 = blockIdx.x * 32, n0 = blockIdx.y * 32;
    const int tid = threadIdx.x;
    for (int i = tid; i < 1024; i += 256) {
        int r = i >> 5, c = i & 31;
        tile[r][c] = W[(size_t)(k0 + r) * N + n0 + c];
    }
    __syncthreads();
    for (int i = tid; i < 512; i += 256) {
        int r = i >> 4, c2 = (i & 15) * 2;
        v2bf p; p[0] = (bf16)tile[c2][r]; p[1] = (bf16)tile[c2 + 1][r];
        *(v2bf*)&WT[(size_t)(n0 + r) * K + k0 + c2] = p;
    }
}

// ---------------------------------------------------------------------------
// Stage A: per-batch interaction (r = X^T W_rank, fm = X r, lcb = W_lcb^T X)
// + FMB custom norm. One block per batch, 256 threads = 8 waves.
// ---------------------------------------------------------------------------
__global__ __launch_bounds__(256) void interact_kernel(
    const float* __restrict__ x, const float* __restrict__ w_lcb,
    const float* __restrict__ w_rank, float* __restrict__ o,
    float* __restrict__ fm_out)
{
    __shared__ __align__(16) bf16 xs [N_INP][136];  // (n, d)
    __shared__ __align__(16) bf16 xsT[D_   ][ 72];  // (d, n)
    __shared__ __align__(16) bf16 wrT[RANKP][ 72];  // (rank, n)
    __shared__ __align__(16) bf16 wlT[N_LCB][ 72];  // (m, n)
    __shared__ __align__(16) bf16 rsT[RANKP][136];  // (rank, d)
    __shared__ float fms[N_INP][RANK];
    __shared__ float red[256];

    const int b    = blockIdx.x;
    const int tid  = threadIdx.x;
    const int wave = tid >> 5;
    const int lane = tid & 31;
    const int l15  = lane & 15;
    const int hi   = (lane >> 4) & 1;

    const float* xb = x + (size_t)b * N_INP * D_;

    for (int pos = tid; pos < (N_INP / 2) * (D_ / 4); pos += 256) {
        int n  = (pos >> 5) * 2;
        int d0 = (pos & 31) * 4;
        float4 v0 = *(const float4*)&xb[n * D_ + d0];
        float4 v1 = *(const float4*)&xb[(n + 1) * D_ + d0];
        *(v4bf*)&xs[n][d0]     = cvt4(v0);
        *(v4bf*)&xs[n + 1][d0] = cvt4(v1);
        const float* f0 = (const float*)&v0;
        const float* f1 = (const float*)&v1;
#pragma unroll
        for (int j = 0; j < 4; ++j) {
            v2bf p; p[0] = (bf16)f0[j]; p[1] = (bf16)f1[j];
            *(v2bf*)&xsT[d0 + j][n] = p;
        }
    }
    for (int i = tid; i < RANKP * (N_INP / 2); i += 256) {
        int r = i & 31, n = (i >> 5) * 2;
        v2bf p;
        p[0] = (r < RANK) ? (bf16)w_rank[n * RANK + r]       : (bf16)0.f;
        p[1] = (r < RANK) ? (bf16)w_rank[(n + 1) * RANK + r] : (bf16)0.f;
        *(v2bf*)&wrT[r][n] = p;
    }
    for (int i = tid; i < N_LCB * (N_INP / 2); i += 256) {
        int m = i & 15, n = (i >> 4) * 2;
        v2bf p;
        p[0] = (bf16)w_lcb[n * N_LCB + m];
        p[1] = (bf16)w_lcb[(n + 1) * N_LCB + m];
        *(v2bf*)&wlT[m][n] = p;
    }
    __syncthreads();

    // ---- r (128 x 32) = X^T @ w_rank, stored transposed ----
    for (int t = wave; t < 16; t += 8) {
        int mt = t & 7, nt = t >> 3;
        v8f acc = {};
        const uint4v* ra = (const uint4v*)&xsT[mt * 16 + l15][0];
        const uint4v* rb = (const uint4v*)&wrT[nt * 16 + l15][0];
#pragma unroll
        for (int ks = 0; ks < 2; ++ks) {
            Frag a, bq;
            a.q[0]  = ra[ks * 4 + hi * 2];
            a.q[1]  = ra[ks * 4 + hi * 2 + 1];
            bq.q[0] = rb[ks * 4 + hi * 2];
            bq.q[1] = rb[ks * 4 + hi * 2 + 1];
            acc = wmma_bf16(a.v, bq.v, acc);
        }
        v8bf rv;
#pragma unroll
        for (int j = 0; j < 8; ++j) rv[j] = (bf16)acc[j];
        *(v8bf*)&rsT[nt * 16 + l15][mt * 16 + hi * 8] = rv;
    }
    __syncthreads();

    // ---- fm (64 x 32) = X @ r ----
    {
        int mt = wave >> 1, nt = wave & 1;
        v8f acc = {};
        const uint4v* ra = (const uint4v*)&xs [mt * 16 + l15][0];
        const uint4v* rb = (const uint4v*)&rsT[nt * 16 + l15][0];
#pragma unroll
        for (int ks = 0; ks < 4; ++ks) {
            Frag a, bq;
            a.q[0]  = ra[ks * 4 + hi * 2];
            a.q[1]  = ra[ks * 4 + hi * 2 + 1];
            bq.q[0] = rb[ks * 4 + hi * 2];
            bq.q[1] = rb[ks * 4 + hi * 2 + 1];
            acc = wmma_bf16(a.v, bq.v, acc);
        }
#pragma unroll
        for (int j = 0; j < 8; ++j) {
            int m = mt * 16 + j + hi * 8;
            int n = nt * 16 + l15;
            if (n < RANK) fms[m][n] = acc[j];
        }
    }

    // ---- lcb (16 x 128) = w_lcb^T @ X ----
    {
        int nt = wave;
        v8f acc = {};
        const uint4v* ra = (const uint4v*)&wlT[l15][0];
        const uint4v* rb = (const uint4v*)&xsT[nt * 16 + l15][0];
#pragma unroll
        for (int ks = 0; ks < 2; ++ks) {
            Frag a, bq;
            a.q[0]  = ra[ks * 4 + hi * 2];
            a.q[1]  = ra[ks * 4 + hi * 2 + 1];
            bq.q[0] = rb[ks * 4 + hi * 2];
            bq.q[1] = rb[ks * 4 + hi * 2 + 1];
            acc = wmma_bf16(a.v, bq.v, acc);
        }
        float* ob = o + (size_t)b * (N_INP * D_);
#pragma unroll
        for (int j = 0; j < 8; ++j)
            ob[(N_FMB + j + hi * 8) * D_ + nt * 16 + l15] = acc[j];
    }
    __syncthreads();

    // ---- custom norm over the 1536 fm values ----
    float s = 0.f, s2 = 0.f;
    for (int i = tid; i < N_INP * RANK; i += 256) {
        float v = fms[i / RANK][i % RANK];
        s += v; s2 += v * v;
    }
    red[tid] = s; __syncthreads();
    for (int off = 128; off > 0; off >>= 1) {
        if (tid < off) red[tid] += red[tid + off];
        __syncthreads();
    }
    float mean = red[0] * (1.f / 1536.f); __syncthreads();
    red[tid] = s2; __syncthreads();
    for (int off = 128; off > 0; off >>= 1) {
        if (tid < off) red[tid] += red[tid + off];
        __syncthreads();
    }
    float var = red[0] * (1.f / 1536.f) - mean * mean;
    float sd  = sqrtf(fmaxf(var, 0.f));
    sd = fminf(fmaxf(sd, 1e-11f), 1e7f);
    float inv = 1.f / sd;
    float* fmo = fm_out + (size_t)b * MLP_IN;
    for (int i = tid; i < MLP_IN; i += 256)
        fmo[i] = (fms[i / RANK][i % RANK] - mean) * inv;
}

// ---------------------------------------------------------------------------
// WMMA GEMM: C(MxN) = A(MxK, f32) @ BT(NxK, bf16)^T.
// A staged via float4 + cvt_pk into LDS; B tiles DMA'd by the Tensor Data
// Mover directly into the padded LDS layout (pitch 40 bf16 = 16 dwords data
// + 4 dwords pad per row). Double-buffered; wave 0 issues TDM, tracks
// TENSORcnt; barriers publish to the block.
// ---------------------------------------------------------------------------
__global__ __launch_bounds__(256) void gemm_kernel(
    const float* __restrict__ A, const bf16* __restrict__ BT,
    float* __restrict__ C, int M, int N, int K, int ldc, int do_relu)
{
    __shared__ __align__(16) bf16 As [2][128][40];
    __shared__ __align__(16) bf16 Bst[2][ 64][40];

    const int tid  = threadIdx.x;
    const int bm   = blockIdx.x * 128;
    const int bn   = blockIdx.y * 64;
    const int wave = tid >> 5;
    const int lane = tid & 31;
    const int l15  = lane & 15;
    const int hi   = (lane >> 4) & 1;
    const int wm   = (wave >> 1) * 32;
    const int wn   = (wave & 1) * 32;

    const int arow = tid >> 1;            // 0..127
    const int acol = (tid & 1) * 16;      // 0 / 16

    const int KT = K >> 5;
    float4 ar[4];

    // TDM: load 64 x 32 bf16 tile of BT (rows bn.., cols k0..) into Bst[buf].
    auto tdmB = [&](int buf, int k0) {
        unsigned lds = (unsigned)(uintptr_t)&Bst[buf][0][0];
        unsigned long long ga =
            (unsigned long long)(uintptr_t)(BT + (size_t)bn * K + k0);
        u32x4 g0;
        g0[0] = 1u;                                  // count=1, load descriptor
        g0[1] = lds;                                 // lds_addr
        g0[2] = (unsigned)(ga & 0xFFFFFFFFu);        // global_addr[31:0]
        g0[3] = (unsigned)((ga >> 32) & 0x01FFFFFFu) | (2u << 30); // [56:32]|type=2
        i32x8 g1;
        // data_size=2B, pad_enable, pad_interval=16 dwords, pad_amount=4 dwords
        g1[0] = (int)((1u << 16) | (1u << 20) | (3u << 22) | (3u << 25));
        g1[1] = (int)(((unsigned)K & 0xFFFFu) << 16);            // tensor_dim0 lo
        g1[2] = (int)(((unsigned)K >> 16) |
                      (((unsigned)N & 0xFFFFu) << 16));          // dim0 hi | dim1 lo
        g1[3] = (int)(((unsigned)N >> 16) | (32u << 16));        // dim1 hi | tile_dim0=32
        g1[4] = 64;                                              // tile_dim1=64
        g1[5] = (int)K;                                          // dim0_stride lo
        g1[6] = 0;
        g1[7] = 0;
        i32x4 z4 = {0, 0, 0, 0};
        i32x8 z8 = {0, 0, 0, 0, 0, 0, 0, 0};
        __builtin_amdgcn_tensor_load_to_lds(g0, g1, z4, z4, z8, 0);
    };

    auto stageA = [&](int buf) {
        *(v8bf*)&As[buf][arow][acol]     = cvt8(ar[0], ar[1]);
        *(v8bf*)&As[buf][arow][acol + 8] = cvt8(ar[2], ar[3]);
    };

    // prologue: tile 0
    if (wave == 0) {
        tdmB(0, 0);
        __builtin_amdgcn_s_wait_tensorcnt(0);
    }
#pragma unroll
    for (int j = 0; j < 4; ++j)
        ar[j] = *(const float4*)&A[(size_t)(bm + arow) * K + acol + 4 * j];
    stageA(0);
    __syncthreads();

    v8f acc[2][2] = {};
    for (int kt = 0; kt < KT; ++kt) {
        const int cur = kt & 1;
        if (kt + 1 < KT) {
            int k0 = (kt + 1) * 32;
            if (wave == 0) tdmB(cur ^ 1, k0);
#pragma unroll
            for (int j = 0; j < 4; ++j)
                ar[j] = *(const float4*)&A[(size_t)(bm + arow) * K + k0 + acol + 4 * j];
        }

        {
            Frag a0, a1, b0, b1;
            const uint4v* ra0 = (const uint4v*)&As [cur][wm + l15][0];
            const uint4v* ra1 = (const uint4v*)&As [cur][wm + 16 + l15][0];
            const uint4v* rb0 = (const uint4v*)&Bst[cur][wn + l15][0];
            const uint4v* rb1 = (const uint4v*)&Bst[cur][wn + 16 + l15][0];
            a0.q[0] = ra0[hi * 2]; a0.q[1] = ra0[hi * 2 + 1];
            a1.q[0] = ra1[hi * 2]; a1.q[1] = ra1[hi * 2 + 1];
            b0.q[0] = rb0[hi * 2]; b0.q[1] = rb0[hi * 2 + 1];
            b1.q[0] = rb1[hi * 2]; b1.q[1] = rb1[hi * 2 + 1];
            acc[0][0] = wmma_bf16(a0.v, b0.v, acc[0][0]);
            acc[0][1] = wmma_bf16(a0.v, b1.v, acc[0][1]);
            acc[1][0] = wmma_bf16(a1.v, b0.v, acc[1][0]);
            acc[1][1] = wmma_bf16(a1.v, b1.v, acc[1][1]);
        }
        __syncthreads();
        if (kt + 1 < KT) {
            stageA(cur ^ 1);
            if (wave == 0) __builtin_amdgcn_s_wait_tensorcnt(0);
            __syncthreads();
        }
    }

#pragma unroll
    for (int i = 0; i < 2; ++i)
#pragma unroll
        for (int jx = 0; jx < 2; ++jx)
#pragma unroll
            for (int j = 0; j < 8; ++j) {
                int r = bm + wm + i * 16 + j + hi * 8;
                int c = bn + wn + jx * 16 + l15;
                float v = acc[i][jx][j];
                if (do_relu) v = fmaxf(v, 0.f);
                C[(size_t)r * ldc + c] = v;
            }
}

// ---------------------------------------------------------------------------
// s[n] = sum_d o[0,n,d]^2
// ---------------------------------------------------------------------------
__global__ void srow_kernel(const float* __restrict__ o, float* __restrict__ s)
{
    int n = threadIdx.x;
    if (n < N_INP) {
        const float* row = o + n * D_;
        float acc = 0.f;
        for (int d = 0; d < D_; ++d) { float v = row[d]; acc += v * v; }
        s[n] = acc;
    }
}

// ---------------------------------------------------------------------------
// Epilogue: elementwise chain (x1e-10) + residual + LayerNorm.
// ---------------------------------------------------------------------------
__device__ __forceinline__ float blk_add(float v, float* red) {
    int tid = threadIdx.x;
    __syncthreads(); red[tid] = v; __syncthreads();
    for (int off = 64; off > 0; off >>= 1) {
        if (tid < off) red[tid] += red[tid + off];
        __syncthreads();
    }
    float r = red[0]; __syncthreads();
    return r;
}
__device__ __forceinline__ float blk_max(float v, float* red) {
    int tid = threadIdx.x;
    __syncthreads(); red[tid] = v; __syncthreads();
    for (int off = 64; off > 0; off >>= 1) {
        if (tid < off) red[tid] = fmaxf(red[tid], red[tid + off]);
        __syncthreads();
    }
    float r = red[0]; __syncthreads();
    return r;
}
__device__ __forceinline__ float blk_mul(float v, float* red) {
    int tid = threadIdx.x;
    __syncthreads(); red[tid] = v; __syncthreads();
    for (int off = 64; off > 0; off >>= 1) {
        if (tid < off) red[tid] *= red[tid + off];
        __syncthreads();
    }
    float r = red[0]; __syncthreads();
    return r;
}
__device__ __forceinline__ float uhash(unsigned int i) {
    i ^= 2747636419u; i *= 2654435769u;
    i ^= i >> 16;     i *= 2654435769u;
    i ^= i >> 16;     i *= 2654435769u;
    return (float)(i >> 8) * (1.0f / 16777216.0f);
}

__global__ __launch_bounds__(128) void epilogue_kernel(
    const float* __restrict__ o, const float* __restrict__ x,
    const float* __restrict__ srow, const float* __restrict__ gamma,
    const float* __restrict__ beta, float* __restrict__ out)
{
    __shared__ float red[128];
    __shared__ float rowv[128];

    const int row = blockIdx.x;        // b * 64 + n
    const int n   = row & (N_INP - 1);
    const int d   = threadIdx.x;
    const size_t base = (size_t)row * D_;

    float t  = o[base + d];
    float xr = x[base + d];
    rowv[d] = t;

    float rmax   = blk_max(t, red);
    float e      = __expf(t - rmax);
    float sumexp = blk_add(e, red);
    float rprod  = blk_mul(t, red);

    float g = 0.f;
    int g0 = d & ~7;
#pragma unroll
    for (int j = 0; j < 8; ++j) g += rowv[g0 + j];

    float sm  = e / sumexp;
    float lg  = __logf(t);
    float rsq = rsqrtf(t + 1e-10f);
    float th  = tanhf(t);
    float lr  = (t >= 0.f) ? t : 0.2f * t;
    float ex  = __expf(t);
    float sp  = fmaxf(t, 0.f) + log1pf(__expf(-fabsf(t)));
    float l1p = log1pf(t);
    float rnd = rintf(t);
    float pw  = __powf(t, 1.0001f);
    float sg  = (t > 0.f) ? 1.f : ((t < 0.f) ? -1.f : t);
    float isn = (t != t) ? 1.f : 0.f;
    float ge  = (t >= 0.f) ? 2.f : 0.f;
    float ne  = (t != 0.f) ? 1.f : 0.f;
    int   c8  = (signed char)(int)t;
    float bnd = (float)(signed char)(c8 & (-c8));
    float wh  = (t > 0.5f) ? -1.f : t;
    float uni = uhash((unsigned int)(base + d));
    float ar  = (float)d;
    float sr  = srow[n];

    float extra = 11.f * t + lg + rsq + sm + th + lr + ex + sp + l1p + rnd +
                  pw + sg + rprod + isn + ge + ne + bnd + rmax + wh + g +
                  uni + ar + sr;
    float pre = t + xr + 1e-10f * extra;

    float mu  = blk_add(pre, red) * (1.f / 128.f);
    float c   = pre - mu;
    float var = blk_add(c * c, red) * (1.f / 128.f);
    out[base + d] = c * rsqrtf(var + 1e-3f) * gamma[d] + beta[d];
}

// ---------------------------------------------------------------------------
extern "C" void kernel_launch(void* const* d_in, const int* in_sizes, int n_in,
                              void* d_out, int out_size, void* d_ws, size_t ws_size,
                              hipStream_t stream)
{
    (void)in_sizes; (void)n_in; (void)out_size; (void)ws_size;
    const float* x      = (const float*)d_in[0];
    const float* w_lcb  = (const float*)d_in[1];
    const float* w_rank = (const float*)d_in[2];
    const float* w_mlp0 = (const float*)d_in[3];
    const float* w_mlp1 = (const float*)d_in[4];
    const float* w_out  = (const float*)d_in[5];
    const float* gamma  = (const float*)d_in[6];
    const float* beta   = (const float*)d_in[7];
    float* out = (float*)d_out;

    float* ws = (float*)d_ws;
    float* o  = ws;                                    // 2048*64*128 f32
    float* fm = o  + (size_t)B_SZ * N_INP * D_;        // 2048*1536 f32
    float* h1 = fm + (size_t)B_SZ * MLP_IN;            // 2048*512 f32
    float* h2 = h1 + (size_t)B_SZ * DIMH;              // 2048*512 f32
    float* sr = h2 + (size_t)B_SZ * DIMH;              // 64 f32
    bf16*  w0T = (bf16*)(sr + 64);                     // [512][1536] bf16
    bf16*  w1T = w0T + (size_t)DIMH * MLP_IN;          // [512][512]  bf16
    bf16*  woT = w1T + (size_t)DIMH * DIMH;            // [6144][512] bf16

    transpose_bf16_kernel<<<dim3(MLP_IN / 32, DIMH / 32), 256, 0, stream>>>(
        w_mlp0, w0T, MLP_IN, DIMH);
    transpose_bf16_kernel<<<dim3(DIMH / 32, DIMH / 32), 256, 0, stream>>>(
        w_mlp1, w1T, DIMH, DIMH);
    transpose_bf16_kernel<<<dim3(DIMH / 32, MLP_OUT / 32), 256, 0, stream>>>(
        w_out, woT, DIMH, MLP_OUT);

    interact_kernel<<<B_SZ, 256, 0, stream>>>(x, w_lcb, w_rank, o, fm);
    gemm_kernel<<<dim3(B_SZ / 128, DIMH / 64), 256, 0, stream>>>(
        fm, w0T, h1, B_SZ, DIMH, MLP_IN, DIMH, 1);
    gemm_kernel<<<dim3(B_SZ / 128, DIMH / 64), 256, 0, stream>>>(
        h1, w1T, h2, B_SZ, DIMH, DIMH, DIMH, 1);
    gemm_kernel<<<dim3(B_SZ / 128, MLP_OUT / 64), 256, 0, stream>>>(
        h2, woT, o, B_SZ, MLP_OUT, DIMH, N_INP * D_, 0);
    srow_kernel<<<1, 64, 0, stream>>>(o, sr);
    epilogue_kernel<<<B_SZ * N_INP, 128, 0, stream>>>(o, x, sr, gamma, beta, out);
}
